// MPNDiff_25254407701135
// MI455X (gfx1250) — compile-verified, matrix operand order checked by reading, not run on hardware
//
#include <hip/hip_runtime.h>
#include <hip/hip_bf16.h>

// ---------------- problem constants (from reference) ----------------
#define N_ATOMS   65536
#define N_BONDS   131072
#define ATOM_FDIM 133
#define BOND_TOTAL 147
#define BOND_FDIM 14
#define HIDDEN    256
#define MAX_NB    6
#define N_MOLS    2048

// padded K sizes (multiples of 32 for WMMA f16 K=32)
#define K_WI 160   // 133 -> 160
#define K_WH 288   // 270 -> 288
#define K_WO 416   // 389 -> 416
#define LDA_AIN 416  // a_input f16 row stride (cols 0..159 atoms, 160..415 a_message)
#define LDA_CAT 288  // concat  f16 row stride (0..255 nei_a, 256..269 nei_b, 270..287 zero)

typedef _Float16 h16;
typedef __attribute__((ext_vector_type(16))) _Float16 v16h;
typedef __attribute__((ext_vector_type(8)))  _Float16 v8h;
typedef __attribute__((ext_vector_type(8)))  float    v8f;

union Frag16 { v16h v; v8h h[2]; };
struct Frags { Frag16 a[4]; Frag16 b[4]; };   // one pipeline stage: 4 M-strips + 4 N-tiles

// ---------------- generic zero ----------------
__global__ void zero_f32_kernel(float* __restrict__ p, int n) {
    int t = blockIdx.x * blockDim.x + threadIdx.x;
    if (t < n) p[t] = 0.0f;
}

// ---------------- weight pack: Wt[n][k] = W[sk][n] (f32->f16, transpose, pad) ----------------
__global__ void pack_w_kernel(const float* __restrict__ W, h16* __restrict__ Wt,
                              int Kp, int Ksrc, int gap_at, int gap_len) {
    int t = blockIdx.x * blockDim.x + threadIdx.x;
    if (t >= HIDDEN * Kp) return;
    int n = t / Kp, k = t - n * Kp;
    float v = 0.0f;
    bool gap = (k >= gap_at) && (k < gap_at + gap_len);
    int sk = (k < gap_at) ? k : (k - gap_len);
    if (!gap && sk < Ksrc) v = W[(size_t)sk * HIDDEN + n];
    Wt[(size_t)n * Kp + k] = (h16)v;
}

// ---------------- pack atom features into a_input cols [0,160) ----------------
__global__ void pack_atoms_kernel(const float* __restrict__ af, h16* __restrict__ a_in) {
    int t = blockIdx.x * blockDim.x + threadIdx.x;      // N_ATOMS*160 threads
    int atom = t / K_WI, c = t - atom * K_WI;
    float v = (c < ATOM_FDIM) ? af[(size_t)atom * ATOM_FDIM + c] : 0.0f;
    a_in[(size_t)atom * LDA_AIN + c] = (h16)v;
}

// ---------------- extract last BOND_FDIM cols of f_bonds -> fb16[N_BONDS][16] ----------------
__global__ void pack_fb_kernel(const float* __restrict__ fb, h16* __restrict__ fb16) {
    int t = blockIdx.x * blockDim.x + threadIdx.x;      // N_BONDS*16 threads
    int b = t >> 4, c = t & 15;
    float v = (c < BOND_FDIM) ? fb[(size_t)b * BOND_TOTAL + (BOND_TOTAL - BOND_FDIM) + c] : 0.0f;
    fb16[(size_t)b * 16 + c] = (h16)v;
}

// ---------------- nei_a_sum: dst[atom][c8..c8+7] = sum_j msg[a2a[atom][j]][c8..] ----------------
__global__ void gather_a_kernel(const h16* __restrict__ msg, const int* __restrict__ a2a,
                                h16* __restrict__ dst, int dstStride) {
    int t = blockIdx.x * blockDim.x + threadIdx.x;      // N_ATOMS*32 threads
    int atom = t >> 5, c8 = (t & 31) * 8;
    float acc[8];
#pragma unroll
    for (int i = 0; i < 8; ++i) acc[i] = 0.0f;
#pragma unroll
    for (int j = 0; j < MAX_NB; ++j) {
        int src = a2a[atom * MAX_NB + j];
        v8h m = *reinterpret_cast<const v8h*>(msg + (size_t)src * HIDDEN + c8);
#pragma unroll
        for (int i = 0; i < 8; ++i) acc[i] += (float)m[i];
    }
    v8h o;
#pragma unroll
    for (int i = 0; i < 8; ++i) o[i] = (h16)acc[i];
    *reinterpret_cast<v8h*>(dst + (size_t)atom * dstStride + c8) = o;
}

// ---------------- nei_b_sum into concat cols [256,288) (>=270 zero pad) ----------------
__global__ void gather_b_kernel(const h16* __restrict__ fb16, const int* __restrict__ a2b,
                                h16* __restrict__ cat) {
    int t = blockIdx.x * blockDim.x + threadIdx.x;      // N_ATOMS*32 threads
    int atom = t >> 5, c = t & 31;
    float v = 0.0f;
    if (c < BOND_FDIM) {
#pragma unroll
        for (int j = 0; j < MAX_NB; ++j) {
            int b = a2b[atom * MAX_NB + j];
            v += (float)fb16[(size_t)b * 16 + c];
        }
    }
    cat[(size_t)atom * LDA_CAT + HIDDEN + c] = (h16)v;
}

// ---------------- counts per molecule ----------------
__global__ void counts_kernel(const int* __restrict__ seg, float* __restrict__ counts) {
    int t = blockIdx.x * blockDim.x + threadIdx.x;
    if (t < N_ATOMS) atomicAdd(&counts[seg[t]], 1.0f);
}

// ---------------- finalize: mean ----------------
__global__ void finalize_kernel(float* __restrict__ out, const float* __restrict__ counts) {
    int t = blockIdx.x * blockDim.x + threadIdx.x;      // N_MOLS*HIDDEN
    int m = t >> 8;
    float c = counts[m];
    out[t] = (c > 0.0f) ? out[t] / c : 0.0f;
}

// ---------------- WMMA GEMM: D[M,256] = A[M,K](f16,row) * Bt[256,K](f16,col-packed) ----------------
// 8 waves as 2(M)x4(N); each wave: 4 M-strips x 4 N-tiles (64x64), block tile 128x256.
// Per k-step: 16 b128 loads feed 16 WMMAs; explicit 2-stage double buffer hides latency.
// K, LDA compile-time so every load uses a constant immediate offset.
// MODE 0 (Wi): out16a=inp16 (pre-relu), out16b=msg16 (relu)
// MODE 1 (Wh): val += inp16 ; out16a=msg16 (relu)
// MODE 2 (Wo): relu, atomicAdd into segout[seg[row]*256+col]
template <int MODE, int K, int LDA>
__global__ __launch_bounds__(256)
void gemm_wmma_kernel(const h16* __restrict__ A,
                      const h16* __restrict__ Bt,
                      const float* __restrict__ bias,
                      h16* __restrict__ out16a,
                      h16* __restrict__ out16b,
                      const h16* __restrict__ addin,
                      const int* __restrict__ segids,
                      float* __restrict__ segout) {
    static_assert((K / 32) % 2 == 1, "double-buffer schedule assumes odd K/32");
    const int lane  = threadIdx.x & 31;
    const int wave  = threadIdx.x >> 5;
    const int waveM = wave >> 2;              // 0..1
    const int waveN = wave & 3;               // 0..3
    const int c     = lane & 15;
    const int hi    = lane >> 4;
    const int rowBlk = blockIdx.x * 128 + waveM * 64;

    // A: lane holds row (strip base + lane&15); lanes<16: K 0..7/16..23, lanes>=16: 8..15/24..31
    const h16* aptr = A + (size_t)(rowBlk + c) * LDA + hi * 8;
    // B: lane holds column (waveN*64 + t*16 + c); pre-transposed so K is contiguous
    const h16* bptr = Bt + (size_t)(waveN * 64 + c) * K + hi * 8;

    v8f acc[4][4] = {};

    auto loadF = [&](Frags& f, int k0) {
#pragma unroll
        for (int r = 0; r < 4; ++r) {
            f.a[r].h[0] = *reinterpret_cast<const v8h*>(aptr + k0 + r * (16 * LDA));
            f.a[r].h[1] = *reinterpret_cast<const v8h*>(aptr + k0 + r * (16 * LDA) + 16);
        }
#pragma unroll
        for (int t = 0; t < 4; ++t) {
            f.b[t].h[0] = *reinterpret_cast<const v8h*>(bptr + k0 + t * (16 * K));
            f.b[t].h[1] = *reinterpret_cast<const v8h*>(bptr + k0 + t * (16 * K) + 16);
        }
    };
    auto mmaF = [&](const Frags& f) {
#pragma unroll
        for (int r = 0; r < 4; ++r)
#pragma unroll
            for (int t = 0; t < 4; ++t)
                acc[r][t] = __builtin_amdgcn_wmma_f32_16x16x32_f16(
                    false, f.a[r].v, false, f.b[t].v, (short)0, acc[r][t], false, false);
    };

    constexpr int NK = K / 32;                // 5, 9, 13 (odd)
    Frags s0, s1;
    loadF(s0, 0);
#pragma unroll
    for (int i = 0; i < NK - 1; i += 2) {
        loadF(s1, (i + 1) * 32);
        mmaF(s0);
        loadF(s0, (i + 2) * 32);
        mmaF(s1);
    }
    mmaF(s0);                                 // final (odd) k-step

    // Epilogue. C/D layout: VGPR s -> row s (lanes 0-15) / row 8+s (lanes 16-31), col = lane&15
#pragma unroll
    for (int r = 0; r < 4; ++r) {
#pragma unroll
        for (int t = 0; t < 4; ++t) {
            const int col = waveN * 64 + t * 16 + c;
            const float b = bias[col];
#pragma unroll
            for (int s = 0; s < 8; ++s) {
                const int row = rowBlk + r * 16 + s + 8 * hi;
                float v = acc[r][t][s] + b;
                if (MODE == 0) {
                    out16a[(size_t)row * HIDDEN + col] = (h16)v;
                    out16b[(size_t)row * HIDDEN + col] = (h16)fmaxf(v, 0.0f);
                } else if (MODE == 1) {
                    v += (float)addin[(size_t)row * HIDDEN + col];
                    out16a[(size_t)row * HIDDEN + col] = (h16)fmaxf(v, 0.0f);
                } else {
                    v = fmaxf(v, 0.0f);
                    atomicAdd(&segout[(size_t)segids[row] * HIDDEN + col], v);
                }
            }
        }
    }
}

// ---------------- host-side launch ----------------
static inline size_t align256(size_t x) { return (x + 255) & ~(size_t)255; }

extern "C" void kernel_launch(void* const* d_in, const int* in_sizes, int n_in,
                              void* d_out, int out_size, void* d_ws, size_t ws_size,
                              hipStream_t stream) {
    (void)in_sizes; (void)n_in; (void)out_size; (void)ws_size;

    const float* atom_features = (const float*)d_in[0];
    const float* f_bonds       = (const float*)d_in[1];
    const int*   a2a           = (const int*)d_in[2];
    const int*   a2b           = (const int*)d_in[3];
    const int*   segment_ids   = (const int*)d_in[4];
    const float* Wi            = (const float*)d_in[5];
    const float* bi            = (const float*)d_in[6];
    const float* Wh            = (const float*)d_in[7];
    const float* bh            = (const float*)d_in[8];
    const float* Wo            = (const float*)d_in[9];
    const float* bo            = (const float*)d_in[10];
    float* out = (float*)d_out;

    // workspace carve-up
    char* ws = (char*)d_ws;
    size_t off = 0;
    auto carve = [&](size_t bytes) { char* p = ws + off; off += align256(bytes); return p; };
    h16*   a_in   = (h16*)  carve((size_t)N_ATOMS * LDA_AIN * sizeof(h16));
    h16*   cat    = (h16*)  carve((size_t)N_ATOMS * LDA_CAT * sizeof(h16));
    h16*   inp16  = (h16*)  carve((size_t)N_ATOMS * HIDDEN * sizeof(h16));
    h16*   msg16  = (h16*)  carve((size_t)N_ATOMS * HIDDEN * sizeof(h16));
    h16*   fb16   = (h16*)  carve((size_t)N_BONDS * 16 * sizeof(h16));
    h16*   Wi_t   = (h16*)  carve((size_t)HIDDEN * K_WI * sizeof(h16));
    h16*   Wh_t   = (h16*)  carve((size_t)HIDDEN * K_WH * sizeof(h16));
    h16*   Wo_t   = (h16*)  carve((size_t)HIDDEN * K_WO * sizeof(h16));
    float* counts = (float*)carve((size_t)N_MOLS * sizeof(float));

    const int BT = 256;

    // init output accumulators
    zero_f32_kernel<<<(N_MOLS * HIDDEN + BT - 1) / BT, BT, 0, stream>>>(out, N_MOLS * HIDDEN);
    zero_f32_kernel<<<(N_MOLS + BT - 1) / BT, BT, 0, stream>>>(counts, N_MOLS);

    // pack weights (transpose + pad + f16)
    pack_w_kernel<<<(HIDDEN * K_WI) / BT, BT, 0, stream>>>(Wi, Wi_t, K_WI, ATOM_FDIM, ATOM_FDIM, K_WI - ATOM_FDIM);
    pack_w_kernel<<<(HIDDEN * K_WH) / BT, BT, 0, stream>>>(Wh, Wh_t, K_WH, HIDDEN + BOND_FDIM, HIDDEN + BOND_FDIM, K_WH - (HIDDEN + BOND_FDIM));
    pack_w_kernel<<<(HIDDEN * K_WO) / BT, BT, 0, stream>>>(Wo, Wo_t, K_WO, ATOM_FDIM + HIDDEN, ATOM_FDIM, K_WI - ATOM_FDIM);

    // pack activations
    pack_atoms_kernel<<<((size_t)N_ATOMS * K_WI) / BT, BT, 0, stream>>>(atom_features, a_in);
    pack_fb_kernel<<<((size_t)N_BONDS * 16) / BT, BT, 0, stream>>>(f_bonds, fb16);

    const int GEMM_GRID = N_ATOMS / 128;

    // inp = atoms @ Wi + bi ; message = relu(inp)
    gemm_wmma_kernel<0, K_WI, LDA_AIN><<<GEMM_GRID, BT, 0, stream>>>(
        a_in, Wi_t, bi, inp16, msg16, nullptr, nullptr, nullptr);

    // depth loop
    for (int d = 0; d < 2; ++d) {
        gather_a_kernel<<<((size_t)N_ATOMS * 32) / BT, BT, 0, stream>>>(msg16, a2a, cat, LDA_CAT);
        gather_b_kernel<<<((size_t)N_ATOMS * 32) / BT, BT, 0, stream>>>(fb16, a2b, cat);
        gemm_wmma_kernel<1, K_WH, LDA_CAT><<<GEMM_GRID, BT, 0, stream>>>(
            cat, Wh_t, bh, msg16, nullptr, inp16, nullptr, nullptr);
    }

    // a_message into a_input cols [160,416)
    gather_a_kernel<<<((size_t)N_ATOMS * 32) / BT, BT, 0, stream>>>(msg16, a2a, a_in + K_WI, LDA_AIN);

    counts_kernel<<<N_ATOMS / BT, BT, 0, stream>>>(segment_ids, counts);

    // atom_hiddens = relu(a_input @ Wo + bo), fused segment-sum via atomics
    gemm_wmma_kernel<2, K_WO, LDA_AIN><<<GEMM_GRID, BT, 0, stream>>>(
        a_in, Wo_t, bo, nullptr, nullptr, nullptr, segment_ids, out);

    finalize_kernel<<<(N_MOLS * HIDDEN) / BT, BT, 0, stream>>>(out, counts);
}
